// HyperGAT_77738908058405
// MI455X (gfx1250) — matrix operand order repaired and checked by reading, not compiled, not run on hardware
//
#include <hip/hip_runtime.h>

typedef __attribute__((ext_vector_type(16))) __bf16    v16bf;
typedef __attribute__((ext_vector_type(8)))  __bf16    v8bf;
typedef __attribute__((ext_vector_type(8)))  float     v8f;
typedef __attribute__((ext_vector_type(4)))  unsigned  uint32x4;
typedef __attribute__((ext_vector_type(8)))  int       int32x8;
typedef __attribute__((ext_vector_type(4)))  int       int32x4;

#define N_NODES 30000
#define N_EDGES 60000
#define A_DEG   6
#define D_DEG   16
#define BATCH   4096
#define F       256
#define S_COLS  7
#define ALPHA   0.2f

// ---------- helpers ----------
__device__ __forceinline__ unsigned short f2bf_bits(float f) {
  unsigned u = __builtin_bit_cast(unsigned, f);
  return (unsigned short)((u + 0x7FFFu + ((u >> 16) & 1u)) >> 16);  // RNE
}
__device__ __forceinline__ float elu_f(float x)   { return x > 0.f ? x : (__expf(x) - 1.f); }
__device__ __forceinline__ float lrelu_f(float x) { return x > 0.f ? x : ALPHA * x; }
__device__ __forceinline__ float wred(float v) {   // wave32 all-lanes sum
#pragma unroll
  for (int o = 16; o > 0; o >>= 1) v += __shfl_xor(v, o, 32);
  return v;
}
__device__ __forceinline__ void store_bf16x8(unsigned short* dst, const float* v) {
  uint4 pk;
  pk.x = (unsigned)f2bf_bits(v[0]) | ((unsigned)f2bf_bits(v[1]) << 16);
  pk.y = (unsigned)f2bf_bits(v[2]) | ((unsigned)f2bf_bits(v[3]) << 16);
  pk.z = (unsigned)f2bf_bits(v[4]) | ((unsigned)f2bf_bits(v[5]) << 16);
  pk.w = (unsigned)f2bf_bits(v[6]) | ((unsigned)f2bf_bits(v[7]) << 16);
  *(uint4*)dst = pk;
}

// ---------- one-time weight pre-pass: fp32 [k][n] -> transposed bf16 bits [n][k] ----------
__global__ __launch_bounds__(256) void convert_weights_kernel(
    const float* __restrict__ W0, const float* __restrict__ W1,
    const float* __restrict__ W2, const float* __restrict__ W3,
    unsigned short* __restrict__ out) {
  const float* Ws[4] = {W0, W1, W2, W3};
  const float* W = Ws[blockIdx.y];
  unsigned short* o = out + (size_t)blockIdx.y * F * F;
  const int idx = blockIdx.x * 256 + threadIdx.x;   // 256 blocks * 256 thr = 65536
  const int k = idx >> 8, n = idx & 255;            // coalesced read over n
  o[n * F + k] = f2bf_bits(W[k * F + n]);
}

// ---------- one-time row conversion: fp32 row-major -> bf16 row-major (8 elems/thread) ----------
__global__ __launch_bounds__(256) void convert_rows_kernel(
    const float* __restrict__ src, unsigned short* __restrict__ dst, int n8) {
  const int i = blockIdx.x * 256 + threadIdx.x;
  if (i >= n8) return;
  const float4* s = (const float4*)src + (size_t)i * 2;
  float4 a = s[0], b = s[1];
  float v[8] = {a.x, a.y, a.z, a.w, b.x, b.y, b.z, b.w};
  store_bf16x8(dst + (size_t)i * 8, v);
}

// ---------- GEMM: Out[M,256] = A16[M,256](bf16) @ W[256,256]; WMMA, f32 acc ----------
// Weight tile (pre-transposed bf16, 128 KB) is DMA'd global->LDS by the Tensor Data Mover.
__global__ __launch_bounds__(256) void gemm256_wmma_bf16(
    const unsigned short* __restrict__ A16, const unsigned short* __restrict__ Wt_g,
    float* __restrict__ Out, int M) {
  __shared__ __align__(32) unsigned short Wt[F * F];  // 128 KB: bf16 bits, [n][k]

  // --- TDM: one wave issues a 2D 256x256 (2B elem) tensor_load_to_lds ---
  if (threadIdx.x < 32) {
    const unsigned long long ga = (unsigned long long)Wt_g;
    // D# group 0: count=1 | lds_addr=0 | global_addr | type=2
    uint32x4 g0 = { 1u, 0u, (unsigned)ga,
                    (unsigned)((ga >> 32) & 0x01FFFFFFu) | 0x80000000u };
    // D# group 1: data_size=1(2B); tensor 256x256, tile 256x256, stride 256
    int32x8 g1 = { (int)0x10000,          // wg_mask=0, data_size=1
                   (int)(256u << 16),     // tensor_dim0[15:0] << 16
                   (int)(256u << 16),     // tensor_dim1[15:0] << 16
                   (int)(256u << 16),     // tile_dim0 << 16
                   256,                   // tile_dim1
                   256,                   // tensor_dim0_stride[31:0]
                   0, 0 };
    int32x4 gz = { 0, 0, 0, 0 };          // groups 2/3: dims 2..4 unused
    // Pass &Wt as an (unused) register operand: this escapes the LDS address
    // into the asm so the "memory" clobber makes LLVM treat Wt as written.
    // (An "=m" operand crashes AMDGPU isel; a bare clobber lets loads of a
    // never-stored LDS global fold to undef and the B-side gets CSE'd away.)
    asm volatile("tensor_load_to_lds %0, %1, %2, %3"
                 :
                 : "s"(g0), "s"(g1), "s"(gz), "s"(gz), "s"((void*)Wt)
                 : "memory");
    __builtin_amdgcn_s_wait_tensorcnt(0);
  }
  __syncthreads();

  const int t    = threadIdx.x;
  const int wave = t >> 5;
  const int lane = t & 31;
  const int r = lane & 15;   // A row / C col within tile
  const int h = lane >> 4;   // lane half
  const int m0 = blockIdx.x * 64 + (wave & 3) * 16;
  if (m0 < M) {
    const int cb = (wave >> 2) * 128;                 // column-block base (8 tiles)
    const unsigned short* arow = A16 + (size_t)(m0 + r) * F;
    const unsigned short* colbase = &Wt[(cb + r) * F + 16 * h];
    v8f acc[8];
#pragma unroll
    for (int j = 0; j < 8; ++j) acc[j] = v8f{0.f, 0.f, 0.f, 0.f, 0.f, 0.f, 0.f, 0.f};

    // ks outer (one live A fragment), j inner over 8 independent accumulators
#pragma unroll
    for (int ks = 0; ks < 8; ++ks) {
      // A fragment: contiguous 16 bf16 [ks*32+16h,+16)
      v16bf afr = *(const v16bf*)(arow + ks * 32 + 16 * h);
#pragma unroll
      for (int j = 0; j < 8; ++j) {
        // B fragment: contiguous 16 bf16 at column (cb+16j+r), k-run ks*32+16h
        v16bf bfr = *(const v16bf*)(colbase + j * 16 * F + ks * 32);
        acc[j] = __builtin_amdgcn_wmma_f32_16x16x32_bf16(
            false, afr, false, bfr, (short)0, acc[j], false, false);
      }
    }
#pragma unroll
    for (int j = 0; j < 8; ++j) {
#pragma unroll
      for (int i = 0; i < 8; ++i)        // C: row = m0 + 8h + i, col = cb + 16j + r
        Out[(size_t)(m0 + 8 * h + i) * F + cb + j * 16 + r] = acc[j][i];
    }
  }
}

// ---------- stage 1: hyperedge attention over member nodes (wave per edge) ----------
// Writes raw new_edge (fp32) and, optionally, bf16(elu(new_edge)) for the next GEMM.
__global__ __launch_bounds__(256) void attn_edge_kernel(
    const float* __restrict__ HE, const float* __restrict__ HN,
    const int* __restrict__ edge_list, const float* __restrict__ ae,
    float* __restrict__ NE, unsigned short* __restrict__ NE16, int E) {
  const int wave = threadIdx.x >> 5, lane = threadIdx.x & 31;
  const int e = blockIdx.x * 8 + wave;
  if (e >= E) return;
  const int c0 = lane * 8;
  float ael[8], aeh[8];
  float base = 0.f;
#pragma unroll
  for (int i = 0; i < 8; ++i) {
    ael[i] = ae[c0 + i];
    aeh[i] = ae[F + c0 + i];
    base += HE[(size_t)e * F + c0 + i] * ael[i];
  }
  base = wred(base);                 // he . ae[:256]
  float gn[A_DEG][8], s[A_DEG];
#pragma unroll
  for (int a = 0; a < A_DEG; ++a) {
    const int id = edge_list[(size_t)e * A_DEG + a];
    const float* row = HN + (size_t)id * F + c0;
    float p = 0.f;
#pragma unroll
    for (int i = 0; i < 8; ++i) { gn[a][i] = row[i]; p += gn[a][i] * aeh[i]; }
    s[a] = base + wred(p);
  }
  float m = -1e30f;
#pragma unroll
  for (int a = 0; a < A_DEG; ++a) { s[a] = lrelu_f(s[a]); m = fmaxf(m, s[a]); }
  float sum = 0.f;
#pragma unroll
  for (int a = 0; a < A_DEG; ++a) { s[a] = __expf(s[a] - m); sum += s[a]; }
  const float inv = 1.f / sum;
  float o[8], oe[8];
#pragma unroll
  for (int i = 0; i < 8; ++i) {
    float v = 0.f;
#pragma unroll
    for (int a = 0; a < A_DEG; ++a) v += s[a] * gn[a][i];
    o[i] = v * inv;
    NE[(size_t)e * F + c0 + i] = o[i];   // raw (pre-ELU) new_edge
  }
  if (NE16) {
#pragma unroll
    for (int i = 0; i < 8; ++i) oe[i] = elu_f(o[i]);
    store_bf16x8(NE16 + (size_t)e * F + c0, oe);
  }
}

// ---------- stage 2: node attention over incident hyperedges (wave per node, 2-pass) ----------
__global__ __launch_bounds__(256) void attn_node_kernel(
    const float* __restrict__ HN, const float* __restrict__ NE,
    const int* __restrict__ node_list, const float* __restrict__ an,
    float* __restrict__ NN, unsigned short* __restrict__ NN16, int Nn) {
  const int wave = threadIdx.x >> 5, lane = threadIdx.x & 31;
  const int n = blockIdx.x * 8 + wave;
  if (n >= Nn) return;
  const int c0 = lane * 8;
  float anh[8];
  float base = 0.f;
#pragma unroll
  for (int i = 0; i < 8; ++i) {
    anh[i] = an[F + c0 + i];
    base += HN[(size_t)n * F + c0 + i] * an[c0 + i];
  }
  base = wred(base);
  int idx[D_DEG];
  float s[D_DEG];
#pragma unroll
  for (int d = 0; d < D_DEG; ++d) {
    idx[d] = node_list[(size_t)n * D_DEG + d];
    const float* row = NE + (size_t)idx[d] * F + c0;
    float p = 0.f;
#pragma unroll
    for (int i = 0; i < 8; ++i) p += row[i] * anh[i];
    s[d] = base + wred(p);
  }
  float m = -1e30f;
#pragma unroll
  for (int d = 0; d < D_DEG; ++d) { s[d] = lrelu_f(s[d]); m = fmaxf(m, s[d]); }
  float sum = 0.f;
#pragma unroll
  for (int d = 0; d < D_DEG; ++d) { s[d] = __expf(s[d] - m); sum += s[d]; }
  const float inv = 1.f / sum;
  float acc[8] = {0.f, 0.f, 0.f, 0.f, 0.f, 0.f, 0.f, 0.f};
#pragma unroll
  for (int d = 0; d < D_DEG; ++d) {        // second pass: L2-resident re-gather
    const float* row = NE + (size_t)idx[d] * F + c0;
#pragma unroll
    for (int i = 0; i < 8; ++i) acc[i] += s[d] * row[i];
  }
  float o[8], oe[8];
#pragma unroll
  for (int i = 0; i < 8; ++i) {
    o[i] = acc[i] * inv;
    NN[(size_t)n * F + c0 + i] = o[i];
  }
  if (NN16) {
#pragma unroll
    for (int i = 0; i < 8; ++i) oe[i] = elu_f(o[i]);
    store_bf16x8(NN16 + (size_t)n * F + c0, oe);
  }
}

// ---------- final assembly (wave per batch row) ----------
__global__ __launch_bounds__(256) void assemble_kernel(
    const int* __restrict__ batch, const float* __restrict__ NN,
    const float* __restrict__ NE, float* __restrict__ out, int Bn) {
  const int wave = threadIdx.x >> 5, lane = threadIdx.x & 31;
  const int b = blockIdx.x * 8 + wave;
  if (b >= Bn) return;
  int bi[S_COLS];
#pragma unroll
  for (int sx = 0; sx < S_COLS; ++sx) bi[sx] = batch[(size_t)b * S_COLS + sx];
  int right = 0;
#pragma unroll
  for (int j = 0; j < S_COLS - 1; ++j)
    if (bi[j] != 0) right = j + 1;
  const int c0 = lane * 8;
  float* ob = out + (size_t)b * S_COLS * F;
#pragma unroll
  for (int i = 0; i < 8; ++i)
    ob[c0 + i] = elu_f(NE[(size_t)bi[0] * F + c0 + i]);   // slot 0 = e2
  for (int sx = 1; sx < S_COLS; ++sx) {
    if (sx < right) {
#pragma unroll
      for (int i = 0; i < 8; ++i)
        ob[sx * F + c0 + i] = elu_f(NN[(size_t)bi[sx] * F + c0 + i]);
    } else {
#pragma unroll
      for (int i = 0; i < 8; ++i) ob[sx * F + c0 + i] = 0.f;
    }
  }
}

// ---------- launcher ----------
extern "C" void kernel_launch(void* const* d_in, const int* in_sizes, int n_in,
                              void* d_out, int out_size, void* d_ws, size_t ws_size,
                              hipStream_t stream) {
  (void)in_sizes; (void)n_in; (void)out_size; (void)ws_size;
  const int*   batch     = (const int*)d_in[0];
  const float* node_embs = (const float*)d_in[1];
  const float* edge_embs = (const float*)d_in[2];
  const int*   edge_list = (const int*)d_in[3];
  const int*   node_list = (const int*)d_in[4];
  const float* Wn1 = (const float*)d_in[5];
  const float* We1 = (const float*)d_in[6];
  const float* ae1 = (const float*)d_in[7];
  const float* an1 = (const float*)d_in[8];
  const float* Wn2 = (const float*)d_in[9];
  const float* We2 = (const float*)d_in[10];
  const float* ae2 = (const float*)d_in[11];
  const float* an2 = (const float*)d_in[12];

  float* HN = (float*)d_ws;                         // [N,256] f32
  float* HE = HN + (size_t)N_NODES * F;             // [E,256] f32
  float* NE = HE + (size_t)N_EDGES * F;             // [E,256] f32
  float* NN = NE + (size_t)N_EDGES * F;             // [N,256] f32
  unsigned short* WB   = (unsigned short*)(NN + (size_t)N_NODES * F); // 4x[256,256] bf16
  unsigned short* X16n = WB + 4 * F * F;            // [N,256] bf16 (embs, then elu(n1))
  unsigned short* X16e = X16n + (size_t)N_NODES * F;// [E,256] bf16 (embs, then elu(e1))

  dim3 blk(256);
  dim3 gW(256, 4);
  dim3 cN(((N_NODES * F / 8) + 255) / 256), cE(((N_EDGES * F / 8) + 255) / 256);
  dim3 gN((N_NODES + 63) / 64), gE((N_EDGES + 63) / 64);
  dim3 aE((N_EDGES + 7) / 8), aN((N_NODES + 7) / 8), aB((BATCH + 7) / 8);

  // one-time conversions: weights (transposed) + input embeddings (row-major)
  convert_weights_kernel<<<gW, blk, 0, stream>>>(Wn1, We1, Wn2, We2, WB);
  convert_rows_kernel<<<cN, blk, 0, stream>>>(node_embs, X16n, N_NODES * F / 8);
  convert_rows_kernel<<<cE, blk, 0, stream>>>(edge_embs, X16e, N_EDGES * F / 8);
  const unsigned short* Wn1b = WB;
  const unsigned short* We1b = WB + 1 * F * F;
  const unsigned short* Wn2b = WB + 2 * F * F;
  const unsigned short* We2b = WB + 3 * F * F;

  // layer 1
  gemm256_wmma_bf16<<<gN, blk, 0, stream>>>(X16n, Wn1b, HN, N_NODES);
  gemm256_wmma_bf16<<<gE, blk, 0, stream>>>(X16e, We1b, HE, N_EDGES);
  attn_edge_kernel<<<aE, blk, 0, stream>>>(HE, HN, edge_list, ae1, NE, X16e, N_EDGES);
  attn_node_kernel<<<aN, blk, 0, stream>>>(HN, NE, node_list, an1, NN, X16n, N_NODES);
  // layer 2 (A inputs already elu'd + bf16 from the attention kernels)
  gemm256_wmma_bf16<<<gN, blk, 0, stream>>>(X16n, Wn2b, HN, N_NODES);
  gemm256_wmma_bf16<<<gE, blk, 0, stream>>>(X16e, We2b, HE, N_EDGES);
  attn_edge_kernel<<<aE, blk, 0, stream>>>(HE, HN, edge_list, ae2, NE, (unsigned short*)nullptr, N_EDGES);
  attn_node_kernel<<<aN, blk, 0, stream>>>(HN, NE, node_list, an2, NN, (unsigned short*)nullptr, N_NODES);
  // final ELU + batch assembly
  assemble_kernel<<<aB, blk, 0, stream>>>(batch, NN, NE, (float*)d_out, BATCH);
}
